// HeteroGNN_12068858102248
// MI455X (gfx1250) — matrix operand order repaired and checked by reading, not compiled, not run on hardware
//
#include <hip/hip_runtime.h>
#include <hip/hip_bf16.h>
#include <math.h>

typedef __attribute__((ext_vector_type(16))) _Float16 v16h;
typedef __attribute__((ext_vector_type(8)))  _Float16 v8h;
typedef __attribute__((ext_vector_type(8)))  float    v8f;

#define TPB 256

static constexpr int kNLig  = 50000;
static constexpr int kNProt = 100000;
static constexpr int kELL = 800000, kEPP = 1600000, kELP = 800000, kEPL = 800000;
static constexpr int kG = 256;

// ---------------------------------------------------------------------------
// helpers
// ---------------------------------------------------------------------------
__device__ __forceinline__ v8f v8f_zero() {
  v8f z;
#pragma unroll
  for (int i = 0; i < 8; ++i) z[i] = 0.0f;
  return z;
}

// Load a 16x32 f16 A-fragment slice for this lane.
// p points at row_base + 8*half; elements 0..7 from p, 8..15 from p+16.
__device__ __forceinline__ v16h load_frag16(const _Float16* p) {
  v8h lo = *reinterpret_cast<const v8h*>(p);
  v8h hi = *reinterpret_cast<const v8h*>(p + 16);
  v16h r;
#pragma unroll
  for (int i = 0; i < 8; ++i) { r[i] = lo[i]; r[i + 8] = hi[i]; }
  return r;
}

__device__ __forceinline__ v8f wmma16(v16h a, v16h b, v8f c) {
  return __builtin_amdgcn_wmma_f32_16x16x32_f16(false, a, false, b, (short)0, c,
                                                false, false);
}

__device__ __forceinline__ void atomic_max_f32(float* addr, float val) {
  if (val >= 0.0f) {
    atomicMax(reinterpret_cast<int*>(addr), __float_as_int(val));
  } else {
    atomicMin(reinterpret_cast<unsigned int*>(addr), __float_as_uint(val));
  }
}

// ---------------------------------------------------------------------------
// elementwise / prep kernels
// ---------------------------------------------------------------------------
__global__ void k_fill_f32(float* p, long long n, float v) {
  long long i = (long long)blockIdx.x * blockDim.x + threadIdx.x;
  long long stride = (long long)gridDim.x * blockDim.x;
  for (; i < n; i += stride) p[i] = v;
}

__global__ void k_count(const int* __restrict__ dst, int E, float* cnt) {
  int e = blockIdx.x * blockDim.x + threadIdx.x;
  if (e < E) atomicAdd(&cnt[dst[e]], 1.0f);
}

// gather x_src[src[e]] (64 f32) and atomic-add into accum[dst[e]]; 4 dims/thread
__global__ void k_scatter(const float* __restrict__ xsrc, const int* __restrict__ src,
                          const int* __restrict__ dst, int E, float* accum) {
  long long t = (long long)blockIdx.x * blockDim.x + threadIdx.x;
  int e = (int)(t >> 4);
  if (e >= E) return;
  int q = ((int)t & 15) * 4;
  const float4 v = *reinterpret_cast<const float4*>(xsrc + (long long)src[e] * 64 + q);
  float* a = accum + (long long)dst[e] * 64 + q;
  atomicAdd(a + 0, v.x); atomicAdd(a + 1, v.y);
  atomicAdd(a + 2, v.z); atomicAdd(a + 3, v.w);
}

__global__ void k_f32_to_f16(const float* __restrict__ in, _Float16* __restrict__ out,
                             long long n) {
  long long t = (long long)blockIdx.x * blockDim.x + threadIdx.x;
  if (t < n) out[t] = (_Float16)in[t];
}

__global__ void k_mean_f16(const float* __restrict__ accum, const float* __restrict__ cnt,
                           _Float16* __restrict__ out, long long n64) {
  long long t = (long long)blockIdx.x * blockDim.x + threadIdx.x;
  if (t < n64) out[t] = (_Float16)(accum[t] / fmaxf(cnt[t >> 6], 1.0f));
}

// d_pl[e][j] = edge_lin.W[j] * attr[e] + edge_lin.b[j]   (f16)
__global__ void k_dpl(const float* __restrict__ attr, const float* __restrict__ W,
                      const float* __restrict__ b, _Float16* __restrict__ out, int E) {
  long long t = (long long)blockIdx.x * blockDim.x + threadIdx.x;
  if (t >= (long long)E * 8) return;
  int e = (int)(t >> 3), j = (int)t & 7;
  out[t] = (_Float16)(W[j] * attr[e] + b[j]);
}

// ---------------------------------------------------------------------------
// weight packing into WMMA B-fragment layout
// packed[((kb*NBT+nb)*32 + lane)*16 + i] = B[kb*32 + 8*(lane>>4) + (i&7) + 16*(i>>3)][nb*16 + (lane&15)]
// ---------------------------------------------------------------------------
// conv B: K=192 (0-63 Wl_a, 64-127 Wl_b, 128-191 Wr_a+Wr_b), N=64 -> kb<6, nb<4
__global__ void k_pack_conv(const float* __restrict__ Wl_a, const float* __restrict__ Wl_b,
                            const float* __restrict__ Wr_a, const float* __restrict__ Wr_b,
                            _Float16* __restrict__ packed) {
  int idx = blockIdx.x * blockDim.x + threadIdx.x;
  if (idx >= 6 * 4 * 32 * 16) return;
  int i = idx & 15, lane = (idx >> 4) & 31, nb = (idx >> 9) & 3, kb = idx >> 11;
  int n = nb * 16 + (lane & 15);
  int k = kb * 32 + 8 * (lane >> 4) + (i & 7) + ((i >> 3) << 4);
  float v;
  if (k < 64)        v = Wl_a[n * 64 + k];
  else if (k < 128)  v = Wl_b[n * 64 + (k - 64)];
  else               v = Wr_a[n * 64 + (k - 128)] + Wr_b[n * 64 + (k - 128)];
  packed[idx] = (_Float16)v;
}

// W1 [64,136] -> K padded to 160, N=64 -> kb<5, nb<4
__global__ void k_pack_w1(const float* __restrict__ W1, _Float16* __restrict__ packed) {
  int idx = blockIdx.x * blockDim.x + threadIdx.x;
  if (idx >= 5 * 4 * 32 * 16) return;
  int i = idx & 15, lane = (idx >> 4) & 31, nb = (idx >> 9) & 3, kb = idx >> 11;
  int n = nb * 16 + (lane & 15);
  int k = kb * 32 + 8 * (lane >> 4) + (i & 7) + ((i >> 3) << 4);
  packed[idx] = (_Float16)((k < 136) ? W1[n * 136 + k] : 0.0f);
}

// W2 [16,64] -> K=64, N=16 -> kb<2, nb==0
__global__ void k_pack_w2(const float* __restrict__ W2, _Float16* __restrict__ packed) {
  int idx = blockIdx.x * blockDim.x + threadIdx.x;
  if (idx >= 2 * 32 * 16) return;
  int i = idx & 15, lane = (idx >> 4) & 31, kb = idx >> 9;
  int n = lane & 15;
  int k = kb * 32 + 8 * (lane >> 4) + (i & 7) + ((i >> 3) << 4);
  packed[idx] = (_Float16)W2[n * 64 + k];
}

// ---------------------------------------------------------------------------
// node update: out = lrelu([meanA|meanB|x] @ Bpacked + bl_a + bl_b)
// one wave per 16-row tile; K=192 (6 k-tiles), N=64 (4 n-tiles) -> 24 WMMA
// ---------------------------------------------------------------------------
__global__ void k_node_update(const _Float16* __restrict__ A1, const _Float16* __restrict__ A2,
                              const _Float16* __restrict__ A3, const _Float16* __restrict__ Bp,
                              const float* __restrict__ b0, const float* __restrict__ b1,
                              float* __restrict__ out_cur, float* __restrict__ out_sum,
                              int ntiles, int first) {
  int w = (int)(((long long)blockIdx.x * blockDim.x + threadIdx.x) >> 5);
  if (w >= ntiles) return;                   // wave-uniform
  int lane = threadIdx.x & 31, m = lane & 15, half = lane >> 4;
  long long row = (long long)w * 16 + m;
  v8f acc[4];
#pragma unroll
  for (int nb = 0; nb < 4; ++nb) acc[nb] = v8f_zero();
  const _Float16* srcs[3] = {A1, A2, A3};
#pragma unroll
  for (int s = 0; s < 3; ++s) {
    const _Float16* arow = srcs[s] + row * 64;
#pragma unroll
    for (int kt = 0; kt < 2; ++kt) {
      v16h a = load_frag16(arow + kt * 32 + 8 * half);
      int kb = s * 2 + kt;
#pragma unroll
      for (int nb = 0; nb < 4; ++nb) {
        v16h b = *reinterpret_cast<const v16h*>(Bp + ((kb * 4 + nb) * 32 + lane) * 16);
        acc[nb] = wmma16(a, b, acc[nb]);
      }
    }
  }
#pragma unroll
  for (int nb = 0; nb < 4; ++nb) {
    int n = nb * 16 + m;
    float bias = b0[n] + b1[n];
#pragma unroll
    for (int r = 0; r < 8; ++r) {
      float v = acc[nb][r] + bias;
      v = (v >= 0.0f) ? v : 0.01f * v;       // leaky relu
      long long idx = ((long long)w * 16 + 8 * half + r) * 64 + n;
      out_cur[idx] = v;
      out_sum[idx] = first ? v : (out_sum[idx] + v);
    }
  }
}

// ---------------------------------------------------------------------------
// edge MLP: hpre = [xl[src]|xp[dst]|d_pl|0] @ W1packed   (K=160, N=64, 20 WMMA)
// ---------------------------------------------------------------------------
__device__ __forceinline__ void edge_hpre(const _Float16* __restrict__ xlh,
                                          const _Float16* __restrict__ xph,
                                          const _Float16* __restrict__ dplrow,
                                          int se, int de, int lane, int half,
                                          const _Float16* __restrict__ W1p, v8f acc[4]) {
#pragma unroll
  for (int kb = 0; kb < 5; ++kb) {
    v16h a = {};
    if (kb < 2)      a = load_frag16(xlh + (long long)se * 64 + kb * 32 + 8 * half);
    else if (kb < 4) a = load_frag16(xph + (long long)de * 64 + (kb - 2) * 32 + 8 * half);
    else {
      if (half == 0) {                       // K 128..135 = d_pl row, rest zero pad
        v8h lo = *reinterpret_cast<const v8h*>(dplrow);
#pragma unroll
        for (int i = 0; i < 8; ++i) a[i] = lo[i];
      }
    }
#pragma unroll
    for (int nb = 0; nb < 4; ++nb) {
      v16h b = *reinterpret_cast<const v16h*>(W1p + ((kb * 4 + nb) * 32 + lane) * 16);
      acc[nb] = wmma16(a, b, acc[nb]);
    }
  }
}

// pass 1: per-column sum / sumsq of hpre (b1 omitted: cancels in BN)
__global__ void k_edge_stats(const _Float16* __restrict__ xlh, const _Float16* __restrict__ xph,
                             const _Float16* __restrict__ dplh,
                             const int* __restrict__ src, const int* __restrict__ dst,
                             const _Float16* __restrict__ W1p,
                             float* __restrict__ colsum, float* __restrict__ colsq, int ntiles) {
  __shared__ float s_sum[64];
  __shared__ float s_sq[64];
  if (threadIdx.x < 64) { s_sum[threadIdx.x] = 0.0f; s_sq[threadIdx.x] = 0.0f; }
  __syncthreads();
  int w = (int)(((long long)blockIdx.x * blockDim.x + threadIdx.x) >> 5);
  int lane = threadIdx.x & 31, m = lane & 15, half = lane >> 4;
  if (w < ntiles) {                          // wave-uniform
    long long e = (long long)w * 16 + m;
    int se = src[e], de = dst[e];
    v8f acc[4];
#pragma unroll
    for (int nb = 0; nb < 4; ++nb) acc[nb] = v8f_zero();
    edge_hpre(xlh, xph, dplh + e * 8, se, de, lane, half, W1p, acc);
#pragma unroll
    for (int nb = 0; nb < 4; ++nb) {
      float s = 0.0f, s2 = 0.0f;
#pragma unroll
      for (int r = 0; r < 8; ++r) { float v = acc[nb][r]; s += v; s2 += v * v; }
      atomicAdd(&s_sum[nb * 16 + m], s);     // ds_add_f32, same column per lane
      atomicAdd(&s_sq[nb * 16 + m], s2);
    }
  }
  __syncthreads();
  if (threadIdx.x < 64) {
    atomicAdd(&colsum[threadIdx.x], s_sum[threadIdx.x]);
    atomicAdd(&colsq[threadIdx.x], s_sq[threadIdx.x]);
  }
}

__global__ void k_bn_finalize(const float* __restrict__ colsum, const float* __restrict__ colsq,
                              const float* __restrict__ gamma, const float* __restrict__ beta,
                              float* __restrict__ scale, float* __restrict__ shift, float invE) {
  int n = threadIdx.x;
  if (n >= 64) return;
  float mu = colsum[n] * invE;
  float var = colsq[n] * invE - mu * mu;
  var = var > 0.0f ? var : 0.0f;
  float sc = gamma[n] * rsqrtf(var + 1e-5f);
  scale[n] = sc;
  shift[n] = beta[n] - mu * sc;
}

// pass 2: hpre -> BN -> relu -> (LDS) -> @W2 + b2 -> m_pl -> tanh(lin_mpl) -> segment ops
__global__ void k_edge_pass2(const _Float16* __restrict__ xlh, const _Float16* __restrict__ xph,
                             const _Float16* __restrict__ dplh,
                             const int* __restrict__ src, const int* __restrict__ dst,
                             const _Float16* __restrict__ W1p, const _Float16* __restrict__ W2p,
                             const float* __restrict__ scale, const float* __restrict__ shift,
                             const float* __restrict__ b2,
                             const float* __restrict__ Wm, const float* __restrict__ bm,
                             const int* __restrict__ batch,
                             float* __restrict__ m_w, float* __restrict__ m_max, int ntiles) {
  __shared__ __align__(32) _Float16 hbuf[8 * 16 * 64];   // per-wave 16x64 f16 h tile
  __shared__ float mbuf[8 * 16 * 16];                    // per-wave 16x16 f32 m_pl tile
  int wslot = threadIdx.x >> 5;
  int w = (int)(((long long)blockIdx.x * blockDim.x + threadIdx.x) >> 5);
  int lane = threadIdx.x & 31, m = lane & 15, half = lane >> 4;
  bool valid = (w < ntiles);                 // wave-uniform
  long long e = (long long)w * 16 + m;
  _Float16* hb = hbuf + wslot * 1024;
  if (valid) {
    int se = src[e], de = dst[e];
    v8f acc[4];
#pragma unroll
    for (int nb = 0; nb < 4; ++nb) acc[nb] = v8f_zero();
    edge_hpre(xlh, xph, dplh + e * 8, se, de, lane, half, W1p, acc);
#pragma unroll
    for (int nb = 0; nb < 4; ++nb) {
      int n = nb * 16 + m;
      float sc = scale[n], sh = shift[n];
#pragma unroll
      for (int r = 0; r < 8; ++r) {
        float v = acc[nb][r] * sc + sh;
        v = v > 0.0f ? v : 0.0f;             // relu
        hb[(8 * half + r) * 64 + n] = (_Float16)v;
      }
    }
  }
  __syncthreads();
  if (valid) {
    v8f mac = v8f_zero();
#pragma unroll
    for (int kt = 0; kt < 2; ++kt) {
      v16h a = load_frag16(hb + m * 64 + kt * 32 + 8 * half);
      v16h b = *reinterpret_cast<const v16h*>(W2p + (kt * 32 + lane) * 16);
      mac = wmma16(a, b, mac);
    }
    float bb = b2[m];                        // n-tile 0: n == m
    float* mb = mbuf + wslot * 256;
#pragma unroll
    for (int r = 0; r < 8; ++r) mb[(8 * half + r) * 16 + m] = mac[r] + bb;
  }
  __syncthreads();
  if (valid && lane < 16) {                  // one lane per edge row
    long long e2 = (long long)w * 16 + lane;
    const float* mb = mbuf + wslot * 256 + lane * 16;
    float mv[16];
#pragma unroll
    for (int j = 0; j < 16; ++j) mv[j] = mb[j];
    int g = batch[src[e2]];
#pragma unroll
    for (int j = 0; j < 16; ++j) {
      float a2 = bm[j];
#pragma unroll
      for (int k = 0; k < 16; ++k) a2 += Wm[j * 16 + k] * mv[k];
      float wj = tanhf(a2);
      atomicAdd(&m_w[g * 16 + j], wj * mv[j]);
      atomic_max_f32(&m_max[g * 16 + j], mv[j]);
    }
  }
}

__global__ void k_output(const float* __restrict__ m_w, const float* __restrict__ m_max,
                         float* __restrict__ out) {
  int t = blockIdx.x * blockDim.x + threadIdx.x;
  if (t >= kG * 16) return;
  int g = t >> 4, j = t & 15;
  out[g * 32 + j] = m_w[t];
  float mx = m_max[t];
  out[g * 32 + 16 + j] = (mx == -INFINITY) ? 0.0f : mx;
}

// ---------------------------------------------------------------------------
// host
// ---------------------------------------------------------------------------
static inline unsigned gridFor(long long n) { return (unsigned)((n + TPB - 1) / TPB); }

extern "C" void kernel_launch(void* const* d_in, const int* in_sizes, int n_in,
                              void* d_out, int out_size, void* d_ws, size_t ws_size,
                              hipStream_t stream) {
  (void)in_sizes; (void)n_in; (void)out_size; (void)ws_size;
  // input layout (pytree / sorted-dict flatten of setup_inputs):
  // 0 x_lig 1 x_prot 2 edge_attr_lp
  // 3+12L + 3t + {0:Wl,1:Wr,2:bl}   t: 0=ll 1=lp 2=pl 3=pp
  // 39 edge_lin.W 40 edge_lin.b 41 lin_mpl.W 42 lin_mpl.b
  // 43 W1 44 W2 45 b1 46 b2 47 beta 48 gamma
  // 49 ei_ll 50 ei_pp 51 ei_lp 52 ei_pl 53 batch_lig
  const float* x_lig  = (const float*)d_in[0];
  const float* x_prot = (const float*)d_in[1];
  const float* e_attr = (const float*)d_in[2];
  auto CW = [&](int L, int t, int which) -> const float* {
    return (const float*)d_in[3 + L * 12 + t * 3 + which];
  };
  const float* edgeW = (const float*)d_in[39];
  const float* edgeB = (const float*)d_in[40];
  const float* linW  = (const float*)d_in[41];
  const float* linB  = (const float*)d_in[42];
  const float* W1    = (const float*)d_in[43];
  const float* W2    = (const float*)d_in[44];
  const float* b2    = (const float*)d_in[46];
  const float* beta  = (const float*)d_in[47];
  const float* gamma = (const float*)d_in[48];
  const int* ei_ll = (const int*)d_in[49];
  const int* ei_pp = (const int*)d_in[50];
  const int* ei_lp = (const int*)d_in[51];
  const int* ei_pl = (const int*)d_in[52];
  const int* batch = (const int*)d_in[53];
  float* out = (float*)d_out;

  // workspace bump allocator (256B aligned)
  char* wsb = (char*)d_ws;
  size_t off = 0;
  auto walloc = [&](size_t bytes) -> char* {
    char* p = wsb + off;
    off += (bytes + 255) & ~(size_t)255;
    return p;
  };
  const size_t L64 = (size_t)kNLig * 64, P64 = (size_t)kNProt * 64;
  float* cnt_ll = (float*)walloc(kNLig * 4);
  float* cnt_pl = (float*)walloc(kNLig * 4);
  float* cnt_pp = (float*)walloc(kNProt * 4);
  float* cnt_lp = (float*)walloc(kNProt * 4);
  float* acc_ll = (float*)walloc(L64 * 4);
  float* acc_pl = (float*)walloc(L64 * 4);
  float* acc_pp = (float*)walloc(P64 * 4);
  float* acc_lp = (float*)walloc(P64 * 4);
  _Float16* mean_ll = (_Float16*)walloc(L64 * 2);
  _Float16* mean_pl = (_Float16*)walloc(L64 * 2);
  _Float16* mean_pp = (_Float16*)walloc(P64 * 2);
  _Float16* mean_lp = (_Float16*)walloc(P64 * 2);
  _Float16* xl_h = (_Float16*)walloc(L64 * 2);
  _Float16* xp_h = (_Float16*)walloc(P64 * 2);
  float* xl_bufA = (float*)walloc(L64 * 4);
  float* xl_bufB = (float*)walloc(L64 * 4);
  float* xp_bufA = (float*)walloc(P64 * 4);
  float* xp_bufB = (float*)walloc(P64 * 4);
  float* xl_sum = (float*)walloc(L64 * 4);
  float* xp_sum = (float*)walloc(P64 * 4);
  _Float16* packL = (_Float16*)walloc(12288 * 2);   // lig conv weights (repacked per layer)
  _Float16* packP = (_Float16*)walloc(12288 * 2);   // prot conv weights
  _Float16* w1p = (_Float16*)walloc(10240 * 2);
  _Float16* w2p = (_Float16*)walloc(1024 * 2);
  _Float16* xls_h = (_Float16*)walloc(L64 * 2);
  _Float16* xps_h = (_Float16*)walloc(P64 * 2);
  _Float16* dpl_h = (_Float16*)walloc((size_t)kELP * 8 * 2);
  float* colsum = (float*)walloc(64 * 4);
  float* colsq  = (float*)walloc(64 * 4);
  float* bnsc   = (float*)walloc(64 * 4);
  float* bnsh   = (float*)walloc(64 * 4);
  float* m_w    = (float*)walloc(kG * 16 * 4);
  float* m_max  = (float*)walloc(kG * 16 * 4);

  // ---- per-dst in-degree counts (layer invariant) ----
  k_fill_f32<<<gridFor(kNLig), TPB, 0, stream>>>(cnt_ll, kNLig, 0.0f);
  k_fill_f32<<<gridFor(kNLig), TPB, 0, stream>>>(cnt_pl, kNLig, 0.0f);
  k_fill_f32<<<gridFor(kNProt), TPB, 0, stream>>>(cnt_pp, kNProt, 0.0f);
  k_fill_f32<<<gridFor(kNProt), TPB, 0, stream>>>(cnt_lp, kNProt, 0.0f);
  k_count<<<gridFor(kELL), TPB, 0, stream>>>(ei_ll + kELL, kELL, cnt_ll);
  k_count<<<gridFor(kEPL), TPB, 0, stream>>>(ei_pl + kEPL, kEPL, cnt_pl);
  k_count<<<gridFor(kEPP), TPB, 0, stream>>>(ei_pp + kEPP, kEPP, cnt_pp);
  k_count<<<gridFor(kELP), TPB, 0, stream>>>(ei_lp + kELP, kELP, cnt_lp);

  // ---- three hetero-SAGE layers ----
  const float* xl_in = x_lig;
  const float* xp_in = x_prot;
  for (int L = 0; L < 3; ++L) {
    k_f32_to_f16<<<gridFor(L64), TPB, 0, stream>>>(xl_in, xl_h, (long long)L64);
    k_f32_to_f16<<<gridFor(P64), TPB, 0, stream>>>(xp_in, xp_h, (long long)P64);
    k_fill_f32<<<gridFor(L64), TPB, 0, stream>>>(acc_ll, (long long)L64, 0.0f);
    k_fill_f32<<<gridFor(L64), TPB, 0, stream>>>(acc_pl, (long long)L64, 0.0f);
    k_fill_f32<<<gridFor(P64), TPB, 0, stream>>>(acc_pp, (long long)P64, 0.0f);
    k_fill_f32<<<gridFor(P64), TPB, 0, stream>>>(acc_lp, (long long)P64, 0.0f);
    k_scatter<<<gridFor((long long)kELL * 16), TPB, 0, stream>>>(xl_in, ei_ll, ei_ll + kELL, kELL, acc_ll);
    k_scatter<<<gridFor((long long)kEPL * 16), TPB, 0, stream>>>(xp_in, ei_pl, ei_pl + kEPL, kEPL, acc_pl);
    k_scatter<<<gridFor((long long)kEPP * 16), TPB, 0, stream>>>(xp_in, ei_pp, ei_pp + kEPP, kEPP, acc_pp);
    k_scatter<<<gridFor((long long)kELP * 16), TPB, 0, stream>>>(xl_in, ei_lp, ei_lp + kELP, kELP, acc_lp);
    k_mean_f16<<<gridFor(L64), TPB, 0, stream>>>(acc_ll, cnt_ll, mean_ll, (long long)L64);
    k_mean_f16<<<gridFor(L64), TPB, 0, stream>>>(acc_pl, cnt_pl, mean_pl, (long long)L64);
    k_mean_f16<<<gridFor(P64), TPB, 0, stream>>>(acc_pp, cnt_pp, mean_pp, (long long)P64);
    k_mean_f16<<<gridFor(P64), TPB, 0, stream>>>(acc_lp, cnt_lp, mean_lp, (long long)P64);
    // lig dst: edge types ll (t=0) and pl (t=2); prot dst: pp (t=3) and lp (t=1)
    k_pack_conv<<<gridFor(12288), TPB, 0, stream>>>(CW(L,0,0), CW(L,2,0), CW(L,0,1), CW(L,2,1), packL);
    k_pack_conv<<<gridFor(12288), TPB, 0, stream>>>(CW(L,3,0), CW(L,1,0), CW(L,3,1), CW(L,1,1), packP);
    float* xl_out = (L & 1) ? xl_bufB : xl_bufA;
    float* xp_out = (L & 1) ? xp_bufB : xp_bufA;
    k_node_update<<<gridFor((long long)(kNLig / 16) * 32), TPB, 0, stream>>>(
        mean_ll, mean_pl, xl_h, packL, CW(L,0,2), CW(L,2,2), xl_out, xl_sum, kNLig / 16, L == 0);
    k_node_update<<<gridFor((long long)(kNProt / 16) * 32), TPB, 0, stream>>>(
        mean_pp, mean_lp, xp_h, packP, CW(L,3,2), CW(L,1,2), xp_out, xp_sum, kNProt / 16, L == 0);
    xl_in = xl_out;
    xp_in = xp_out;
  }

  // ---- edge MLP + readout ----
  k_f32_to_f16<<<gridFor(L64), TPB, 0, stream>>>(xl_sum, xls_h, (long long)L64);
  k_f32_to_f16<<<gridFor(P64), TPB, 0, stream>>>(xp_sum, xps_h, (long long)P64);
  k_dpl<<<gridFor((long long)kELP * 8), TPB, 0, stream>>>(e_attr, edgeW, edgeB, dpl_h, kELP);
  k_pack_w1<<<gridFor(10240), TPB, 0, stream>>>(W1, w1p);
  k_pack_w2<<<gridFor(1024), TPB, 0, stream>>>(W2, w2p);
  k_fill_f32<<<1, TPB, 0, stream>>>(colsum, 64, 0.0f);
  k_fill_f32<<<1, TPB, 0, stream>>>(colsq, 64, 0.0f);
  const int etiles = kELP / 16;  // 50000
  k_edge_stats<<<gridFor((long long)etiles * 32), TPB, 0, stream>>>(
      xls_h, xps_h, dpl_h, ei_lp, ei_lp + kELP, w1p, colsum, colsq, etiles);
  k_bn_finalize<<<1, 64, 0, stream>>>(colsum, colsq, gamma, beta, bnsc, bnsh, 1.0f / kELP);
  k_fill_f32<<<gridFor(kG * 16), TPB, 0, stream>>>(m_w, kG * 16, 0.0f);
  k_fill_f32<<<gridFor(kG * 16), TPB, 0, stream>>>(m_max, kG * 16, -INFINITY);
  k_edge_pass2<<<gridFor((long long)etiles * 32), TPB, 0, stream>>>(
      xls_h, xps_h, dpl_h, ei_lp, ei_lp + kELP, w1p, w2p, bnsc, bnsh, b2,
      linW, linB, batch, m_w, m_max, etiles);
  k_output<<<gridFor(kG * 16), TPB, 0, stream>>>(m_w, m_max, out);
}
